// Time_info_aggregation_86698209837471
// MI455X (gfx1250) — matrix operand order repaired and checked by reading, not compiled, not run on hardware
//
#include <hip/hip_runtime.h>
#include <hip/hip_bf16.h>
#include <stdint.h>

// ---------------------------------------------------------------------------
// Mamba forward for MI455X (gfx1250, wave32).
// Heavy GEMMs: LDS-staged (pair-packed B), register-blocked
// v_wmma_f32_16x16x32_bf16 (fp32 accumulate), compile-time strides so all
// staging loads use immediate offsets. Fragments are fetched with
// ds_load_b128 only. Recurrence / norms / activations stay fp32 VALU.
// ---------------------------------------------------------------------------

typedef __attribute__((ext_vector_type(16))) __bf16        v16bf;
typedef __attribute__((ext_vector_type(8)))  float         v8f;
typedef __attribute__((ext_vector_type(8)))  unsigned int  v8u;
typedef __attribute__((ext_vector_type(4)))  unsigned int  v4u;

#define NLAYER 6
#define DMODEL 256
#define DINNER 512
#define BSZ    16
#define SEQL   1024
#define NTOK   (BSZ * SEQL)      // 16384
#define DTRANK 16
#define DSTATE 16
#define NCOLS  (DTRANK + 2 * DSTATE)   // 48

__device__ __forceinline__ unsigned short f2bf(float f) {
    unsigned u = __builtin_bit_cast(unsigned, f);
    unsigned r = (u + 0x7FFFu + ((u >> 16) & 1u)) >> 16;
    return (unsigned short)r;
}
__device__ __forceinline__ float siluf(float x) {
    return x / (1.f + __expf(-x));
}
__device__ __forceinline__ float softplusf(float x) {
    return (x > 20.f) ? x : __logf(1.f + __expf(x));
}

// ---------------------------------------------------------------------------
// f32 -> bf16 elementwise conversion (weights per layer)
// ---------------------------------------------------------------------------
__global__ void k_f2bf(const float* __restrict__ src, unsigned short* __restrict__ dst, int n) {
    int i = blockIdx.x * blockDim.x + threadIdx.x;
    if (i < n) dst[i] = f2bf(src[i]);
}

// ---------------------------------------------------------------------------
// Fused residual update + RMSNorm.
// mode 0: resid = hidden_in            (layer 0: hidden_in = x)
// mode 1: resid = resid + hidden_in
// Then o = resid * rsqrt(mean(resid^2)+eps) * w; emit bf16 and/or f32.
// Block = 256 threads = DMODEL, one token per block.
// ---------------------------------------------------------------------------
__global__ void k_rmsnorm(float* __restrict__ resid,
                          const float* __restrict__ hidden_in,
                          const float* __restrict__ w,
                          unsigned short* __restrict__ out_bf,
                          float* __restrict__ out_f32,
                          int mode) {
    const size_t tok = blockIdx.x;
    const int i = threadIdx.x;
    const size_t off = tok * DMODEL + i;
    float v;
    if (mode == 0) {
        v = hidden_in[off];
    } else {
        v = resid[off] + hidden_in[off];
    }
    resid[off] = v;

    float ss = v * v;
    #pragma unroll
    for (int m = 16; m >= 1; m >>= 1) ss += __shfl_xor(ss, m);
    __shared__ float partial[8];
    if ((threadIdx.x & 31) == 0) partial[threadIdx.x >> 5] = ss;
    __syncthreads();
    float tot = 0.f;
    #pragma unroll
    for (int j = 0; j < 8; ++j) tot += partial[j];
    const float scale = rsqrtf(tot * (1.f / DMODEL) + 1e-5f);
    const float o = v * scale * w[i];
    if (out_bf)  out_bf[off]  = f2bf(o);
    if (out_f32) out_f32[off] = o;
}

// ---------------------------------------------------------------------------
// BF16 WMMA GEMM: C[M,N] (f32) = A[M,K] (bf16 rm) * B[K,N] (bf16 rm).
// N, K are template constants -> all staging addresses are base + immediate.
// Block = 256 thr = 8 waves arranged 2(M) x 4(N).
// Block tile = 32M x 256N; each wave: 16M x 64N = 4 WMMA tiles (A reused 4x).
//
// LDS layouts:
//   lds_a : A chunk 32 rows x 32 k, row-major (lane fragment = 2x b128).
//   lds_bp: B chunk pre-packed into WMMA pair format. For column n, the 16
//           u32 pairs (B[2kp][n] | B[2kp+1][n]<<16), kp=0..15, live at
//           lds_bp[n*16 + (kp half-block XOR-swizzled by n&1)]. A lane's B
//           sub-fragment is one contiguous 32 B v8u read -> two ds_load_b128.
//
// Staging: global reads lane-coalesced; LDS writes ds_store_b128. Next
// weight chunk prefetched (global_prefetch_b8).
// grid = (N/256, M/32); dims tile exactly. A/B fragments use contiguous k = half*16 + element; C/D per ISA.
// ---------------------------------------------------------------------------
template <int N, int K>
__global__ void k_wmma_gemm(const unsigned short* __restrict__ A,
                            const unsigned short* __restrict__ B,
                            float* __restrict__ C) {
    __shared__ unsigned short lds_a[32 * 32];        //  2 KB
    __shared__ unsigned int   lds_bp[256 * 16];      // 16 KB (packed pairs)

    const int tid  = threadIdx.x;
    const int lane = tid & 31;
    const int wave = tid >> 5;
    const int wm   = wave >> 2;          // 0..1 : which 16-row half
    const int wn   = wave & 3;           // 0..3 : which 64-col quarter
    const int row0 = blockIdx.y * 32;
    const int nblk = blockIdx.x * 256;
    const int m    = lane & 15;
    const int half = lane >> 4;

    v8f acc[4] = {};

    // Per-thread invariant staging bases (strides are compile-time).
    const int ar  = tid >> 3;                 // A stage row
    const int akk = (tid & 7) * 4;            // A stage k offset
    const unsigned short* aSrc = A + (size_t)(row0 + ar) * K + akk;

    for (int kb = 0; kb < K; kb += 32) {
        __syncthreads();
        // ---- stage A chunk: 32 rows x 32 k  (8 B per thread, b64) ----
        *(uint64_t*)(&lds_a[ar * 32 + akk]) = *(const uint64_t*)(aSrc + kb);

        // ---- stage B chunk: 32 k x 256 n, packed into u32 pairs ----
        // 512 slots of 8 dwords; thread handles 2 slots. Global u16 reads
        // are coalesced across lanes (consecutive n); writes are b128.
        #pragma unroll
        for (int it = 0; it < 2; ++it) {
            const int slot = tid + it * 256;
            const int n    = slot & 255;         // column within block tile
            const int kh   = slot >> 8;          // 0..1 : which 16-k half
            const int bh   = kh ^ (n & 1);       // swizzled half-block
            const unsigned short* bSrc = B + (size_t)(kb + kh * 16) * N + nblk + n;
            v4u w0, w1;
            #pragma unroll
            for (int j = 0; j < 8; ++j) {
                const unsigned lo = bSrc[(2 * j    ) * N];   // immediate offsets
                const unsigned hi = bSrc[(2 * j + 1) * N];
                const unsigned wv = lo | (hi << 16);
                if (j < 4) w0[j] = wv; else w1[j - 4] = wv;
            }
            *(v4u*)(&lds_bp[n * 16 + bh * 8    ]) = w0;
            *(v4u*)(&lds_bp[n * 16 + bh * 8 + 4]) = w1;
        }
        // prefetch next weight chunk into cache while we compute
        if (kb + 32 < K) {
            __builtin_prefetch(B + (size_t)(kb + 32 + (tid >> 5)) * N + nblk + (tid & 31) * 8, 0, 1);
        }
        __syncthreads();

        // ---- A fragment (16x32 bf16): lane m holds row wm*16+m ----
        // ka = half*16 + v*2 -> one contiguous 32 B run (2x b128)
        v8u au;
        #pragma unroll
        for (int v = 0; v < 8; ++v) {
            const int g  = v >> 2;
            const int vv = v & 3;
            const int ka = half * 16 + v * 2;
            au[v] = *(const unsigned int*)(&lds_a[(wm * 16 + m) * 32 + ka]);
        }
        const v16bf a = __builtin_bit_cast(v16bf, au);

        // ---- 4 x B fragments: one contiguous v8u (2x b128) each ----
        #pragma unroll
        for (int t = 0; t < 4; ++t) {
            const int nn = wn * 64 + t * 16 + m;
            const int eh = half ^ (nn & 1);
            const v8u bu = *(const v8u*)(&lds_bp[nn * 16 + eh * 8]);
            const v16bf b = __builtin_bit_cast(v16bf, bu);
            acc[t] = __builtin_amdgcn_wmma_f32_16x16x32_bf16(
                /*neg_a=*/false, a, /*neg_b=*/false, b,
                /*c_mod=*/(short)0, acc[t], /*reuse_a=*/false, /*reuse_b=*/false);
        }
    }

    // ---- store: C/D layout VGPR v -> M = v + half*8, N = lane&15 ----
    #pragma unroll
    for (int t = 0; t < 4; ++t) {
        #pragma unroll
        for (int v = 0; v < 8; ++v) {
            const int row = row0 + wm * 16 + v + half * 8;
            const int col = nblk + wn * 64 + t * 16 + m;
            C[(size_t)row * N + col] = acc[t][v];
        }
    }
}

// ---------------------------------------------------------------------------
// Depthwise causal conv (width 4) + bias + silu. u = xz[:, 0:512].
// One thread per (b,t,d).
// ---------------------------------------------------------------------------
__global__ void k_conv_silu(const float* __restrict__ xz,
                            const float* __restrict__ cw,
                            const float* __restrict__ cb,
                            float* __restrict__ uc) {
    const size_t idx = (size_t)blockIdx.x * blockDim.x + threadIdx.x;
    if (idx >= (size_t)NTOK * DINNER) return;
    const int d  = (int)(idx % DINNER);
    const size_t bt = idx / DINNER;
    const int t  = (int)(bt % SEQL);
    const size_t b = bt / SEQL;

    float acc = cb[d];
    #pragma unroll
    for (int j = 0; j < 4; ++j) {
        const int tt = t - 3 + j;
        if (tt >= 0)
            acc += xz[((b * SEQL + tt) * (size_t)(2 * DINNER)) + d] * cw[d * 4 + j];
    }
    uc[bt * DINNER + d] = siluf(acc);
}

// ---------------------------------------------------------------------------
// x_dbl = uc @ xp_w  (K=512, N=48). One token per block; LDS-stage the row,
// coalesced weight column reads.
// ---------------------------------------------------------------------------
__global__ void k_xproj(const float* __restrict__ uc,
                        const float* __restrict__ xpw,
                        float* __restrict__ x_dbl) {
    __shared__ float srow[DINNER];
    const size_t tok = blockIdx.x;
    for (int k = threadIdx.x; k < DINNER; k += blockDim.x)
        srow[k] = uc[tok * DINNER + k];
    __syncthreads();
    const int c = threadIdx.x;
    if (c < NCOLS) {
        float acc = 0.f;
        for (int k = 0; k < DINNER; ++k)
            acc += srow[k] * xpw[k * NCOLS + c];
        x_dbl[tok * NCOLS + c] = acc;
    }
}

// ---------------------------------------------------------------------------
// Selective scan. 16 lanes per channel (one per state s), wave32 xor-shuffle
// reductions. Computes dt = softplus(dt_r @ dt_w + dt_b), the h recurrence,
// y = (sum_s h*C + u*D) * silu(z), emits bf16 for the out-proj GEMM.
// Block = 256 thr = 16 channels; grid = B * (DINNER/16) = 512.
// ---------------------------------------------------------------------------
__global__ void k_scan(const float* __restrict__ x_dbl,
                       const float* __restrict__ xz,      // z at col offset DINNER
                       const float* __restrict__ uc,
                       const float* __restrict__ dtw,     // (16, DINNER)
                       const float* __restrict__ dtb,
                       const float* __restrict__ A_log,   // (DINNER, 16)
                       const float* __restrict__ Dp,
                       unsigned short* __restrict__ y_bf) {
    const int s     = threadIdx.x & 15;
    const int chl   = threadIdx.x >> 4;
    const int chunk = blockIdx.x % (DINNER / 16);
    const int b     = blockIdx.x / (DINNER / 16);
    const int d     = chunk * 16 + chl;

    const float A   = -__expf(A_log[d * DSTATE + s]);
    const float w_s = dtw[s * DINNER + d];
    const float bia = dtb[d];
    const float Dd  = Dp[d];

    const float* xd = x_dbl + (size_t)b * SEQL * NCOLS;
    const float* ub = uc    + (size_t)b * SEQL * DINNER;
    const float* zb = xz    + (size_t)b * SEQL * (2 * DINNER) + DINNER;

    float h = 0.f;
    for (int t = 0; t < SEQL; ++t) {
        const float* row = xd + (size_t)t * NCOLS;
        // dt = softplus( sum_r x_dbl[:,r]*dt_w[r,d] + dt_b[d] ), rank r = lane s
        float dtp = row[s] * w_s;
        #pragma unroll
        for (int mm = 8; mm >= 1; mm >>= 1) dtp += __shfl_xor(dtp, mm);
        const float dt = softplusf(dtp + bia);

        const float u  = ub[(size_t)t * DINNER + d];
        const float Bt = row[DTRANK + s];
        const float Ct = row[DTRANK + DSTATE + s];

        h = __expf(dt * A) * h + (dt * u) * Bt;

        float yp = h * Ct;
        #pragma unroll
        for (int mm = 8; mm >= 1; mm >>= 1) yp += __shfl_xor(yp, mm);

        if (s == 0) {
            const float z = zb[(size_t)t * (2 * DINNER) + d];
            const float y = (yp + u * Dd) * siluf(z);
            y_bf[((size_t)b * SEQL + t) * DINNER + d] = f2bf(y);
        }
    }
}

// ---------------------------------------------------------------------------
// Host side
// ---------------------------------------------------------------------------
static inline size_t alignUp(size_t x, size_t a) { return (x + a - 1) & ~(a - 1); }

extern "C" void kernel_launch(void* const* d_in, const int* in_sizes, int n_in,
                              void* d_out, int out_size, void* d_ws, size_t ws_size,
                              hipStream_t stream) {
    (void)in_sizes; (void)n_in; (void)out_size; (void)ws_size;

    const float* x       = (const float*)d_in[0];   // (16,1024,256)
    const float* norm_w  = (const float*)d_in[1];   // (6,256)
    const float* in_w    = (const float*)d_in[2];   // (6,256,1024)
    const float* conv_w  = (const float*)d_in[3];   // (6,512,4)
    const float* conv_b  = (const float*)d_in[4];   // (6,512)
    const float* xp_w    = (const float*)d_in[5];   // (6,512,48)
    const float* dt_w    = (const float*)d_in[6];   // (6,16,512)
    const float* dt_b    = (const float*)d_in[7];   // (6,512)
    const float* A_log   = (const float*)d_in[8];   // (6,512,16)
    const float* Dp      = (const float*)d_in[9];   // (6,512)
    const float* out_w   = (const float*)d_in[10];  // (6,512,256)
    const float* normf_w = (const float*)d_in[11];  // (256)
    float* out           = (float*)d_out;

    // Workspace carve-up
    uint8_t* ws = (uint8_t*)d_ws;
    size_t off = 0;
    float*          resid   = (float*)(ws + off);          off = alignUp(off + (size_t)NTOK * DMODEL * 4, 256);
    float*          hidden  = (float*)(ws + off);          off = alignUp(off + (size_t)NTOK * DMODEL * 4, 256);
    unsigned short* xn_bf   = (unsigned short*)(ws + off); off = alignUp(off + (size_t)NTOK * DMODEL * 2, 256);
    float*          xz      = (float*)(ws + off);          off = alignUp(off + (size_t)NTOK * 2 * DINNER * 4, 256);
    float*          ucb     = (float*)(ws + off);          off = alignUp(off + (size_t)NTOK * DINNER * 4, 256);
    float*          xdbl    = (float*)(ws + off);          off = alignUp(off + (size_t)NTOK * NCOLS * 4, 256);
    unsigned short* y_bf    = (unsigned short*)(ws + off); off = alignUp(off + (size_t)NTOK * DINNER * 2, 256);
    unsigned short* inw_bf  = (unsigned short*)(ws + off); off = alignUp(off + (size_t)DMODEL * 2 * DINNER * 2, 256);
    unsigned short* outw_bf = (unsigned short*)(ws + off); off = alignUp(off + (size_t)DINNER * DMODEL * 2, 256);

    for (int i = 0; i < NLAYER; ++i) {
        // bf16 weight copies for this layer
        {
            int n1 = DMODEL * 2 * DINNER;   // 262144
            k_f2bf<<<(n1 + 255) / 256, 256, 0, stream>>>(in_w + (size_t)i * n1, inw_bf, n1);
            int n2 = DINNER * DMODEL;       // 131072
            k_f2bf<<<(n2 + 255) / 256, 256, 0, stream>>>(out_w + (size_t)i * n2, outw_bf, n2);
        }
        // residual update + rmsnorm -> bf16 activations
        k_rmsnorm<<<NTOK, DMODEL, 0, stream>>>(
            resid, (i == 0) ? x : hidden, norm_w + (size_t)i * DMODEL,
            xn_bf, (float*)nullptr, (i == 0) ? 0 : 1);

        // xz = xn @ in_w    (M=16384, N=1024, K=256)
        {
            dim3 grid((2 * DINNER) / 256, NTOK / 32);
            k_wmma_gemm<2 * DINNER, DMODEL><<<grid, 256, 0, stream>>>(xn_bf, inw_bf, xz);
        }
        // depthwise conv + silu
        {
            size_t tot = (size_t)NTOK * DINNER;
            k_conv_silu<<<(unsigned)((tot + 255) / 256), 256, 0, stream>>>(
                xz, conv_w + (size_t)i * DINNER * 4, conv_b + (size_t)i * DINNER, ucb);
        }
        // x_dbl = uc @ xp_w
        k_xproj<<<NTOK, 64, 0, stream>>>(ucb, xp_w + (size_t)i * DINNER * NCOLS, xdbl);

        // selective scan (+ dt proj, + silu(z) gate) -> y_bf
        k_scan<<<BSZ * (DINNER / 16), 256, 0, stream>>>(
            xdbl, xz, ucb,
            dt_w + (size_t)i * DTRANK * DINNER, dt_b + (size_t)i * DINNER,
            A_log + (size_t)i * DINNER * DSTATE, Dp + (size_t)i * DINNER, y_bf);

        // hidden = y @ out_w   (M=16384, N=256, K=512)
        {
            dim3 grid(DMODEL / 256, NTOK / 32);
            k_wmma_gemm<DMODEL, DINNER><<<grid, 256, 0, stream>>>(y_bf, outw_bf, hidden);
        }
    }

    // final: resid += hidden; out = rmsnorm(resid, norm_f_w)
    k_rmsnorm<<<NTOK, DMODEL, 0, stream>>>(
        resid, hidden, normf_w, (unsigned short*)nullptr, out, 1);
}
